// S4Backbone_21414706938534
// MI455X (gfx1250) — matrix operand order repaired and checked by reading, not compiled, not run on hardware
//
#include <hip/hip_runtime.h>
#include <hip/hip_bf16.h>
#include <cstdint>
#include <math.h>

// Problem constants (reference: B=8, T=2048, D=256, N=64, L=4)
#define B_  8
#define T_  2048
#define D_  256
#define N_  64
#define L_  4
#define M_  (B_*T_)          // 16384 rows
#define LN_EPS 1e-5f

typedef __attribute__((ext_vector_type(16))) __bf16 v16bf;
typedef __attribute__((ext_vector_type(8)))  float  v8f;

// ---------------------------------------------------------------- wave utils
__device__ __forceinline__ int lane_id() {
  return (int)__builtin_amdgcn_mbcnt_lo(~0u, 0u);   // wave32: 0..31
}
__device__ __forceinline__ float wave_bcast(float v, int src) {
  return __int_as_float(__builtin_amdgcn_ds_bpermute(src << 2, __float_as_int(v)));
}
__device__ __forceinline__ float wave_allsum(float v) {
  int lane = lane_id();
#pragma unroll
  for (int m = 16; m >= 1; m >>= 1)
    v += __int_as_float(__builtin_amdgcn_ds_bpermute(((lane ^ m) << 2), __float_as_int(v)));
  return v;
}

// ------------------------------------------------------- fp32 -> bf16 convert
__global__ void f32_to_bf16_kernel(const float* __restrict__ src,
                                   __hip_bfloat16* __restrict__ dst, int n) {
  int i = blockIdx.x * 256 + threadIdx.x;
  if (i < n) dst[i] = __float2bfloat16(src[i]);
}

// ------------------------------------------------- S4D parameter precompute
// Abar = exp(-exp(A_log)*dt), CB = Cm * (Abar-1)/(A*sqrt(N))
__global__ void s4_precompute_kernel(const float* __restrict__ A_log,
                                     const float* __restrict__ Cm,
                                     const float* __restrict__ log_dt,
                                     float* __restrict__ Abar,
                                     float* __restrict__ CB) {
  int i = blockIdx.x * 256 + threadIdx.x;     // over L*D*N
  if (i >= L_ * D_ * N_) return;
  int ld = i / N_;                            // l*D + d
  float dt = expf(log_dt[ld]);
  float A  = -expf(A_log[i]);
  float ab = expf(A * dt);
  float Bb = (ab - 1.0f) / (A * 8.0f);        // sqrt(N)=8
  Abar[i] = ab;
  CB[i]   = Cm[i] * Bb;
}

// ----------------------------------------------------------------- LayerNorm
// one wave per row of D=256; OUT_BF16=1 -> bf16 (GEMM A-matrix), else fp32
template<int OUT_BF16>
__global__ __launch_bounds__(256)
void ln_kernel(const float* __restrict__ x, const float* __restrict__ g,
               const float* __restrict__ b, void* __restrict__ out) {
  int wid = threadIdx.x >> 5, lane = threadIdx.x & 31;
  int row = blockIdx.x * 8 + wid;
  const float* xr = x + (size_t)row * D_;
  float v[8], s = 0.f, s2 = 0.f;
#pragma unroll
  for (int i = 0; i < 8; ++i) {
    v[i] = xr[lane + 32 * i];
    s += v[i]; s2 += v[i] * v[i];
  }
  s  = wave_allsum(s);
  s2 = wave_allsum(s2);
  float mean = s * (1.0f / D_);
  float var  = s2 * (1.0f / D_) - mean * mean;
  float rstd = rsqrtf(var + LN_EPS);
#pragma unroll
  for (int i = 0; i < 8; ++i) {
    int col = lane + 32 * i;
    float o = (v[i] - mean) * rstd * g[col] + b[col];
    if (OUT_BF16) ((__hip_bfloat16*)out)[(size_t)row * D_ + col] = __float2bfloat16(o);
    else          ((float*)out)[(size_t)row * D_ + col] = o;
  }
}

// ---------------------------------------------------------- S4D recurrence
// one wave per (b,d) channel; 2 of the N=64 states per lane; sequential in T
// with 32-step chunks: chunked u load, ds_bpermute broadcast per step.
__global__ __launch_bounds__(256)
void s4_scan_kernel(const float* __restrict__ u, const float* __restrict__ Abar,
                    const float* __restrict__ CB, const float* __restrict__ Dp,
                    __hip_bfloat16* __restrict__ ybf) {
  int wid = threadIdx.x >> 5, lane = threadIdx.x & 31;
  int ch = blockIdx.x * 8 + wid;          // 0 .. B*D-1
  int b = ch / D_, d = ch % D_;
  int n0 = lane * 2;
  float a0 = Abar[d * N_ + n0], a1 = Abar[d * N_ + n0 + 1];
  float c0 = CB  [d * N_ + n0], c1 = CB  [d * N_ + n0 + 1];
  float dp = Dp[d];
  const float* ub = u + ((size_t)b * T_) * D_ + d;
  __hip_bfloat16* yb = ybf + ((size_t)b * T_) * D_ + d;
  float x0 = 0.f, x1 = 0.f;
  for (int t0 = 0; t0 < T_; t0 += 32) {
    float uv = ub[(size_t)(t0 + lane) * D_];
    float yreg = 0.f;
#pragma unroll
    for (int j = 0; j < 32; ++j) {
      float ut = wave_bcast(uv, j);
      x0 = fmaf(a0, x0, ut);
      x1 = fmaf(a1, x1, ut);
      float sv = fmaf(c0, x0, c1 * x1);
      sv = wave_allsum(sv);
      float yt = sv + dp * ut;
      yreg = (lane == j) ? yt : yreg;
    }
    yb[(size_t)(t0 + lane) * D_] = __float2bfloat16(yreg);
  }
}

// ------------------------------------------------------------- WMMA GEMM
// C[M,N] = epilogue( A_bf16[M,K] * W_bf16[K,N] + bias[N] )
// ACT: 0 none, 1 exact GELU. RES: add R fp32. OUTF: fp32 out. OUTB: bf16 out.
// Block: 256 thr = 8 waves (2Mx4N), wave tile 32x32 = 2x2 WMMA 16x16x32 bf16.
// Double-buffered software pipeline:
//   A tile: global_load_async_to_lds_b128 issued one step ahead (ASYNCcnt;
//           in-order completion => s_wait_asynccnt 1 means current tile done)
//   B tile: global_load_b128 into regs one step ahead, ds_store transpose
//           into [n][k] LDS layout so fragments are two ds_load_b128 each.
template<int ACT, int RES, int OUTF, int OUTB>
__global__ __launch_bounds__(256)
void gemm_wmma_kernel(const __hip_bfloat16* __restrict__ A,
                      const __hip_bfloat16* __restrict__ W,
                      const float* __restrict__ bias,
                      const float* __restrict__ R,
                      float* __restrict__ Cf,
                      __hip_bfloat16* __restrict__ Cb,
                      int M, int N, int K) {
  // per buffer: A tile [64][32] bf16 = 4096B at +0, B tile [128][32] bf16 = 8192B at +4096
  __shared__ __align__(16) unsigned char smem[2 * 12288];

  const int tid  = threadIdx.x;
  const int lane = tid & 31, wid = tid >> 5;
  const int waveM = wid >> 2, waveN = wid & 3;     // 2 x 4 wave grid
  const int h = lane >> 4, lm = lane & 15;
  const int blockM = blockIdx.y * 64, blockN = blockIdx.x * 128;

  v8f acc[2][2];
#pragma unroll
  for (int mi = 0; mi < 2; ++mi)
#pragma unroll
    for (int ni = 0; ni < 2; ++ni)
      acc[mi][ni] = {};

  // A staging: 64 rows x 64B; each thread one async b128 (16B)
  const int ar = tid >> 2;                 // row 0..63
  const int ac = (tid & 3) * 16;           // byte-in-row

  auto ldA_async = [&](int ks, int buf) {
    const unsigned char* gA = (const unsigned char*)A
        + (size_t)(blockM + ar) * ((size_t)K * 2) + (size_t)ks * 2 + ac;
    unsigned lds = (unsigned)(buf * 12288 + ar * 64 + ac);   // smem at LDS offset 0
    unsigned long long ga64 = (unsigned long long)(size_t)gA;
    asm volatile("global_load_async_to_lds_b128 %0, %1, off"
                 :: "v"(lds), "v"(ga64) : "memory");
  };
  auto ldB = [&](int ks, uint4* q) {
#pragma unroll
    for (int s = 0; s < 2; ++s) {
      int idx = tid + s * 256;
      int k = idx >> 4;                    // 0..31
      int n = (idx & 15) * 8;              // 0..120 step 8
      q[s] = *(const uint4*)((const unsigned char*)W +
               ((size_t)(ks + k) * N + blockN + n) * 2);
    }
  };
  auto stB = [&](const uint4* q, int buf) {
    unsigned short* bt = (unsigned short*)(smem + buf * 12288 + 4096);
#pragma unroll
    for (int s = 0; s < 2; ++s) {
      int idx = tid + s * 256;
      int k = idx >> 4;
      int n = (idx & 15) * 8;
      union { uint4 q_; unsigned short u16[8]; } v;
      v.q_ = q[s];
#pragma unroll
      for (int j = 0; j < 8; ++j) bt[(n + j) * 32 + k] = v.u16[j];
    }
  };

  const int nk = K >> 5;
  uint4 breg[2], bnext[2];

  // pipeline prologue: tile 0 in flight
  ldB(0, breg);
  ldA_async(0, 0);

  for (int i = 0; i < nk; ++i) {
    const int cur = i & 1;
    // finish staging current B tile (regs loaded one step ago -> LDS)
    stB(breg, cur);
    // prefetch next tile (dup of last tile into dead buffer on final iter
    // keeps the "<=1 outstanding async => current done" invariant)
    const int inx = (i + 1 < nk) ? (i + 1) : i;
    ldA_async(inx << 5, cur ^ 1);
    ldB(inx << 5, bnext);
    // current A async copy complete (in-order; only the prefetch may remain)
    asm volatile("s_wait_asynccnt 0x1" ::: "memory");
    __syncthreads();

    // ---- fragment loads per documented CDNA5 WMMA VGPR layouts ----
    const unsigned char* aT = smem + cur * 12288;
    const unsigned char* bT = aT + 4096;
    union AF { uint4 q[2]; v16bf v; };
    AF af[2], bf[2];
#pragma unroll
    for (int mi = 0; mi < 2; ++mi) {
      int row = waveM * 32 + mi * 16 + lm;
      af[mi].q[0] = *(const uint4*)(aT + row * 64 + 16 * h);       // K pairs 8h+2v
      af[mi].q[1] = *(const uint4*)(aT + row * 64 + 32 + 16 * h);  // K 16+8h+2v
    }
#pragma unroll
    for (int ni = 0; ni < 2; ++ni) {
      int rowb = waveN * 32 + ni * 16 + lm;
      bf[ni].q[0] = *(const uint4*)(bT + rowb * 64 + 16 * h);      // K 8h+0..7
      bf[ni].q[1] = *(const uint4*)(bT + rowb * 64 + 32 + 16 * h); // K 16+8h+0..7
    }
#pragma unroll
    for (int mi = 0; mi < 2; ++mi)
#pragma unroll
      for (int ni = 0; ni < 2; ++ni)
        acc[mi][ni] = __builtin_amdgcn_wmma_f32_16x16x32_bf16(
            false, af[mi].v, false, bf[ni].v, (short)0, acc[mi][ni], false, false);
    __syncthreads();   // all reads of `cur` done before it is overwritten

    breg[0] = bnext[0];
    breg[1] = bnext[1];
  }

  // ---- epilogue: bias (+GELU) (+residual) -> fp32 / bf16 ----
#pragma unroll
  for (int mi = 0; mi < 2; ++mi) {
#pragma unroll
    for (int ni = 0; ni < 2; ++ni) {
      int n = blockN + waveN * 32 + ni * 16 + lm;
      float bn = bias[n];
      size_t off = (size_t)(blockM + waveM * 32 + mi * 16 + 8 * h) * (size_t)N + n;
#pragma unroll
      for (int r = 0; r < 8; ++r) {
        float v = acc[mi][ni][r] + bn;
        if (ACT == 1) v = 0.5f * v * (1.0f + erff(v * 0.70710678118654752f));
        if (RES)  v += R[off];
        if (OUTF) Cf[off] = v;
        if (OUTB) Cb[off] = __float2bfloat16(v);
        off += (size_t)N;
      }
    }
  }
}

// ---------------------------------------------------------------- launcher
extern "C" void kernel_launch(void* const* d_in, const int* in_sizes, int n_in,
                              void* d_out, int out_size, void* d_ws, size_t ws_size,
                              hipStream_t stream) {
  (void)in_sizes; (void)n_in; (void)out_size; (void)ws_size;
  const float* seq    = (const float*)d_in[0];
  const float* Wi     = (const float*)d_in[1];
  const float* bi     = (const float*)d_in[2];
  const float* A_log  = (const float*)d_in[3];
  const float* Cm     = (const float*)d_in[4];
  const float* Dp     = (const float*)d_in[5];
  const float* log_dt = (const float*)d_in[6];
  const float* Wo     = (const float*)d_in[7];
  const float* bo     = (const float*)d_in[8];
  const float* ln1g   = (const float*)d_in[9];
  const float* ln1b   = (const float*)d_in[10];
  const float* ln2g   = (const float*)d_in[11];
  const float* ln2b   = (const float*)d_in[12];
  const float* W1     = (const float*)d_in[13];
  const float* b1     = (const float*)d_in[14];
  const float* W2     = (const float*)d_in[15];
  const float* b2     = (const float*)d_in[16];
  const float* lnfg   = (const float*)d_in[17];
  const float* lnfb   = (const float*)d_in[18];

  // workspace carve-out (256B aligned)
  char* p = (char*)d_ws;
  auto carve = [&](size_t bytes) -> void* {
    void* r = (void*)p; p += (bytes + 255) & ~(size_t)255; return r;
  };
  float*          x    = (float*)         carve((size_t)M_ * D_ * 4);       // activations
  __hip_bfloat16* hbf  = (__hip_bfloat16*)carve((size_t)M_ * D_ * 2);       // LN out (GEMM A)
  float*          u    = (float*)         carve((size_t)M_ * D_ * 4);       // S4 input
  __hip_bfloat16* ybf  = (__hip_bfloat16*)carve((size_t)M_ * D_ * 2);       // S4 output
  __hip_bfloat16* mbf  = (__hip_bfloat16*)carve((size_t)M_ * 4 * D_ * 2);   // GELU out
  __hip_bfloat16* Wibf = (__hip_bfloat16*)carve((size_t)L_ * D_ * D_ * 2);
  __hip_bfloat16* Wobf = (__hip_bfloat16*)carve((size_t)L_ * D_ * D_ * 2);
  __hip_bfloat16* W1bf = (__hip_bfloat16*)carve((size_t)L_ * D_ * 4 * D_ * 2);
  __hip_bfloat16* W2bf = (__hip_bfloat16*)carve((size_t)L_ * 4 * D_ * D_ * 2);
  float*          Abar = (float*)         carve((size_t)L_ * D_ * N_ * 4);
  float*          CBs  = (float*)         carve((size_t)L_ * D_ * N_ * 4);

  // one-time (per launch) weight conversion + S4 parameter precompute
  {
    int n1 = L_ * D_ * D_, n2 = L_ * D_ * 4 * D_;
    f32_to_bf16_kernel<<<(n1 + 255) / 256, 256, 0, stream>>>(Wi, Wibf, n1);
    f32_to_bf16_kernel<<<(n1 + 255) / 256, 256, 0, stream>>>(Wo, Wobf, n1);
    f32_to_bf16_kernel<<<(n2 + 255) / 256, 256, 0, stream>>>(W1, W1bf, n2);
    f32_to_bf16_kernel<<<(n2 + 255) / 256, 256, 0, stream>>>(W2, W2bf, n2);
    int n3 = L_ * D_ * N_;
    s4_precompute_kernel<<<(n3 + 255) / 256, 256, 0, stream>>>(A_log, Cm, log_dt, Abar, CBs);
  }
  hipMemcpyAsync(x, seq, (size_t)M_ * D_ * 4, hipMemcpyDeviceToDevice, stream);

  const dim3 blk(256);
  const dim3 gLN(M_ / 8);
  const dim3 gGemmD(D_ / 128, M_ / 64);        // N=256
  const dim3 gGemm4D(4 * D_ / 128, M_ / 64);   // N=1024
  const dim3 gScan((B_ * D_) / 8);

  for (int l = 0; l < L_; ++l) {
    // h = LN1(x)  [bf16]
    ln_kernel<1><<<gLN, blk, 0, stream>>>(x, ln1g + l * D_, ln1b + l * D_, hbf);
    // u = h @ Wi + bi  [fp32]
    gemm_wmma_kernel<0,0,1,0><<<gGemmD, blk, 0, stream>>>(
        hbf, Wibf + (size_t)l * D_ * D_, bi + l * D_, nullptr, u, nullptr, M_, D_, D_);
    // y = S4 scan(u) + Dp*u  [bf16, (B,T,D)]
    s4_scan_kernel<<<gScan, blk, 0, stream>>>(
        u, Abar + (size_t)l * D_ * N_, CBs + (size_t)l * D_ * N_, Dp + l * D_, ybf);
    // x = x + y @ Wo + bo
    gemm_wmma_kernel<0,1,1,0><<<gGemmD, blk, 0, stream>>>(
        ybf, Wobf + (size_t)l * D_ * D_, bo + l * D_, x, x, nullptr, M_, D_, D_);
    // h2 = LN2(x)  [bf16]
    ln_kernel<1><<<gLN, blk, 0, stream>>>(x, ln2g + l * D_, ln2b + l * D_, hbf);
    // g = gelu(h2 @ W1 + b1)  [bf16]
    gemm_wmma_kernel<1,0,0,1><<<gGemm4D, blk, 0, stream>>>(
        hbf, W1bf + (size_t)l * D_ * 4 * D_, b1 + l * 4 * D_, nullptr, nullptr, mbf,
        M_, 4 * D_, D_);
    // x = x + g @ W2 + b2
    gemm_wmma_kernel<0,1,1,0><<<gGemmD, blk, 0, stream>>>(
        mbf, W2bf + (size_t)l * 4 * D_ * D_, b2 + l * D_, x, x, nullptr, M_, D_, 4 * D_);
  }
  // out = LNf(x)  [fp32]
  ln_kernel<0><<<gLN, blk, 0, stream>>>(x, lnfg, lnfb, d_out);
}
